// Attention_40870908788792
// MI455X (gfx1250) — matrix operand order repaired and checked by reading, not compiled, not run on hardware
//
#include <hip/hip_runtime.h>
#include <hip/hip_fp16.h>

typedef _Float16 v16h __attribute__((ext_vector_type(16)));
typedef float    v8f  __attribute__((ext_vector_type(8)));
typedef unsigned u32x4 __attribute__((ext_vector_type(4)));
typedef int      i32x8 __attribute__((ext_vector_type(8)));
typedef int      i32x4 __attribute__((ext_vector_type(4)));

#define NTOK 49
#define NPAD 64
#define CDIM 512
#define DH   32
#define NH   16

__device__ __forceinline__ v8f wmma_f16(v16h a, v16h b, v8f c) {
  // D = A(16x32 f16) * B(32x16 f16) + C(16x16 f32)
  return __builtin_amdgcn_wmma_f32_16x16x32_f16(false, a, false, b, (short)0, c,
                                                false, false);
}

// one tensor (q, k or v): depthwise 3x3 conv + relu + LayerNorm -> f16 LDS.
// dst element (token n, channel c) stored at dst[n*rs + c*cs] so V can be
// written transposed ([c][n]) for contiguous WMMA B-fragment loads.
__device__ __forceinline__ void qkv_one(const float* __restrict__ xbuf,
                                        const float* __restrict__ cw,
                                        const float* __restrict__ cb,
                                        const float* __restrict__ lg,
                                        const float* __restrict__ lb,
                                        _Float16* __restrict__ dst, int rs,
                                        int cs, float scale, int wv, int lane) {
  for (int n = wv; n < NTOK; n += 8) {
    const int i = n / 7, j = n % 7;
    float yv[16];
    float s = 0.f, s2 = 0.f;
#pragma unroll
    for (int u = 0; u < 16; ++u) {
      const int c = lane + 32 * u;
      float acc = cb[c];
#pragma unroll
      for (int di = 0; di < 3; ++di) {
        const int ii = i + di - 1;
        if (ii < 0 || ii >= 7) continue;
#pragma unroll
        for (int dj = 0; dj < 3; ++dj) {
          const int jj = j + dj - 1;
          if (jj < 0 || jj >= 7) continue;
          acc += cw[c * 9 + di * 3 + dj] * xbuf[(ii * 7 + jj) * CDIM + c];
        }
      }
      acc = fmaxf(acc, 0.f);
      yv[u] = acc;
      s += acc;
      s2 += acc * acc;
    }
#pragma unroll
    for (int off = 16; off > 0; off >>= 1) {
      s  += __shfl_xor(s,  off, 32);
      s2 += __shfl_xor(s2, off, 32);
    }
    const float mu   = s * (1.f / 512.f);
    const float var  = s2 * (1.f / 512.f) - mu * mu;
    const float rstd = rsqrtf(var + 1e-5f);
#pragma unroll
    for (int u = 0; u < 16; ++u) {
      const int c = lane + 32 * u;
      dst[n * rs + c * cs] =
          (_Float16)(((yv[u] - mu) * rstd * lg[c] + lb[c]) * scale);
    }
  }
}

__global__ __launch_bounds__(256) void attn_win_kernel(
    const float* __restrict__ x, const float* __restrict__ qw,
    const float* __restrict__ qb, const float* __restrict__ qg,
    const float* __restrict__ qbeta, const float* __restrict__ kw,
    const float* __restrict__ kb_, const float* __restrict__ kg,
    const float* __restrict__ kbeta, const float* __restrict__ vw,
    const float* __restrict__ vb, const float* __restrict__ vg,
    const float* __restrict__ vbeta, const float* __restrict__ relb,
    _Float16* __restrict__ attnX) {
  extern __shared__ char smem[];
  float*    xbuf = (float*)smem;                        // 100352 B @ offset 0
  _Float16* qs   = (_Float16*)(smem + 100352);          // 64 x 512 f16 [n][c]
  _Float16* ks   = (_Float16*)(smem + 100352 + 65536);  // 64 x 512 f16 [n][c]
  _Float16* vs   = (_Float16*)(smem + 100352 + 131072); // 512 x 64 f16 [c][n]

  const int win    = blockIdx.x;
  const int tid    = threadIdx.x;
  const int lane   = tid & 31;
  const int wv     = tid >> 5;
  const int lane16 = lane >> 4;  // which half of the wave
  const int lanelo = lane & 15;

  // ---- stage window input (49x512 f32) into LDS with the Tensor Data Mover
  if (wv == 0) {
    const unsigned long long ga =
        (unsigned long long)(x + (size_t)win * NTOK * CDIM);
    u32x4 g0;
    g0[0] = 1u;                                   // count=1, user descriptor
    g0[1] = 0u;                                   // lds_addr = 0 (xbuf)
    g0[2] = (unsigned)(ga & 0xffffffffu);         // global_addr[31:0]
    g0[3] = (unsigned)((ga >> 32) & 0x01ffffffu)  // global_addr[56:32]
            | 0x80000000u;                        // type=2 ("image")
    i32x8 g1;
    g1[0] = (int)(2u << 16);    // data_size = 4 bytes
    g1[1] = (int)(512u << 16);  // tensor_dim0 = 512 (low 16 -> bits[63:48])
    g1[2] = (int)(49u << 16);   // tensor_dim1 = 49  (low 16 -> bits[95:80])
    g1[3] = (int)(512u << 16);  // tile_dim0 = 512   (bits[127:112])
    g1[4] = 49;                 // tile_dim1 = 49    (bits[143:128])
    g1[5] = 512;                // tensor_dim0_stride[31:0] = 512
    g1[6] = 0;                  // stride0 hi / stride1 lo
    g1[7] = 0;
    i32x4 g2 = {0, 0, 0, 0};    // 2D tensor: groups 2/3 unused
    i32x4 g3 = {0, 0, 0, 0};
    i32x8 g4 = {0, 0, 0, 0, 0, 0, 0, 0};  // extra group (6-arg toolchain form)
    __builtin_amdgcn_tensor_load_to_lds(g0, g1, g2, g3, g4, 0);
    __builtin_amdgcn_s_wait_tensorcnt(0);
  }

  // zero-pad token rows 49..63 of q/k (row-major) and v (transposed)
  for (int t = tid; t < (NPAD - NTOK) * CDIM; t += 256) {
    const int r = NTOK + t / CDIM, c = t % CDIM;
    qs[r * CDIM + c] = (_Float16)0.f;
    ks[r * CDIM + c] = (_Float16)0.f;
    vs[c * NPAD + r] = (_Float16)0.f;
  }
  __syncthreads();  // TDM data + pads visible to all waves

  const float scale = 0.17677669529663687f;  // 32^-0.5 folded into q
  qkv_one(xbuf, qw, qb,  qg, qbeta, qs, CDIM, 1, scale, wv, lane);
  qkv_one(xbuf, kw, kb_, kg, kbeta, ks, CDIM, 1, 1.f,   wv, lane);
  qkv_one(xbuf, vw, vb,  vg, vbeta, vs, 1, NPAD, 1.f,   wv, lane);
  __syncthreads();  // q/k/v complete; x region free for P scratch

  _Float16* pbuf = (_Float16*)(smem + wv * 8192);  // wave-private 64x64 f16

  for (int hh = 0; hh < 2; ++hh) {
    const int h     = wv * 2 + hh;
    const int cbase = h * DH;

    // ---- sim = Q * K^T : 4x4 tiles of 16x16, single K-step (K=32=dh) ----
    v8f acc[4][4];
#pragma unroll
    for (int mi = 0; mi < 4; ++mi) {
      v16h a;
      const _Float16* ap = qs + (16 * mi + lanelo) * CDIM + cbase + 16 * lane16;
#pragma unroll
      for (int e = 0; e < 16; ++e) a[e] = ap[e];
#pragma unroll
      for (int nj = 0; nj < 4; ++nj) {
        v16h b;
        const _Float16* bp = ks + (16 * nj + lanelo) * CDIM + cbase + 16 * lane16;
#pragma unroll
        for (int e = 0; e < 16; ++e) b[e] = bp[e];
        v8f z = {};
        acc[mi][nj] = wmma_f16(a, b, z);
      }
    }

    // ---- +bias, softmax over j (rows live across lanes of a 16-lane half) --
#pragma unroll
    for (int mi = 0; mi < 4; ++mi) {
#pragma unroll
      for (int g = 0; g < 8; ++g) {
        const int i = 16 * mi + g + 8 * lane16;
        float rowv[4];
#pragma unroll
        for (int nj = 0; nj < 4; ++nj) {
          const int j = 16 * nj + lanelo;
          float sv;
          if (i >= NTOK)      sv = 0.f;      // dead rows, kept finite
          else if (j >= NTOK) sv = -1e30f;   // masked key padding
          else {
            const int rel = (i / 7 - j / 7 + 6) * 13 + (i % 7 - j % 7 + 6);
            sv = acc[mi][nj][g] + relb[rel * NH + h];
          }
          rowv[nj] = sv;
        }
        float m4 = fmaxf(fmaxf(rowv[0], rowv[1]), fmaxf(rowv[2], rowv[3]));
#pragma unroll
        for (int off = 8; off > 0; off >>= 1)
          m4 = fmaxf(m4, __shfl_xor(m4, off, 16));
        float es = 0.f;
#pragma unroll
        for (int nj = 0; nj < 4; ++nj) {
          rowv[nj] = __expf(rowv[nj] - m4);
          es += rowv[nj];
        }
#pragma unroll
        for (int off = 8; off > 0; off >>= 1) es += __shfl_xor(es, off, 16);
        const float inv = 1.f / es;
#pragma unroll
        for (int nj = 0; nj < 4; ++nj)
          pbuf[i * NPAD + 16 * nj + lanelo] = (_Float16)(rowv[nj] * inv);
      }
    }

    // ---- out = P(49x64) * V(64x32) : 4x2 tiles, 2 K-steps of 32 ----
#pragma unroll
    for (int mi = 0; mi < 4; ++mi) {
#pragma unroll
      for (int nd = 0; nd < 2; ++nd) {
        v8f o = {};
        const int cc = cbase + 16 * nd + lanelo;
#pragma unroll
        for (int kk = 0; kk < 2; ++kk) {
          v16h a;
          const _Float16* ap =
              pbuf + (16 * mi + lanelo) * NPAD + 32 * kk + 16 * lane16;
#pragma unroll
          for (int e = 0; e < 16; ++e) a[e] = ap[e];
          v16h b;  // V transposed in LDS: contiguous along tokens
          const _Float16* bp = vs + cc * NPAD + 32 * kk + 16 * lane16;
#pragma unroll
          for (int e = 0; e < 16; ++e) b[e] = bp[e];
          o = wmma_f16(a, b, o);
        }
#pragma unroll
        for (int g = 0; g < 8; ++g) {
          const int n = 16 * mi + g + 8 * lane16;
          if (n < NTOK)
            attnX[((size_t)win * NTOK + n) * CDIM + cc] = (_Float16)o[g];
        }
      }
    }
  }
}

// Y(100352x512 f32) = X(f16) * W^T ; W is (512 out, 512 in) f32, cvt on load
__global__ __launch_bounds__(256) void proj_kernel(
    const _Float16* __restrict__ X, const float* __restrict__ W,
    float* __restrict__ Y) {
  __shared__ __align__(16) _Float16 Xs[128 * 32];   // [row][k]
  __shared__ __align__(16) _Float16 Wsh[128 * 32];  // [outcol][k]
  const int tid    = threadIdx.x;
  const int lane   = tid & 31;
  const int wv     = tid >> 5;
  const int lane16 = lane >> 4;
  const int lanelo = lane & 15;
  const int wm     = wv & 3;   // 4 waves along M (32 rows each)
  const int wn     = wv >> 2;  // 2 waves along N (64 cols each)
  const size_t m0  = (size_t)blockIdx.x * 128;  // 784 * 128 = 100352 exactly
  const int    n0  = blockIdx.y * 128;          // 4 * 128 = 512 exactly

  v8f zero = {};
  v8f acc[2][4];
#pragma unroll
  for (int mi = 0; mi < 2; ++mi)
#pragma unroll
    for (int nj = 0; nj < 4; ++nj) acc[mi][nj] = zero;

  const int lrow = tid >> 1;        // 0..127
  const int lseg = (tid & 1) * 16;  // 0 or 16

  // LDS byte offset of this thread's X-tile destination (generic ptr low bits
  // == LDS offset within the workgroup allocation)
  const unsigned xs_off =
      (unsigned)(unsigned long long)(void*)&Xs[lrow * 32 + lseg];

  for (int k0 = 0; k0 < 512; k0 += 32) {
    // X tile: 128 rows x 32 halves, HBM -> LDS via async-to-LDS (ASYNCcnt)
    const unsigned long long ga =
        (unsigned long long)(X + (m0 + lrow) * 512 + k0 + lseg);
    asm volatile("global_load_async_to_lds_b128 %0, %1, off" ::"v"(xs_off),
                 "v"(ga)
                 : "memory");
    asm volatile("global_load_async_to_lds_b128 %0, %1, off offset:16" ::"v"(
                     xs_off),
                 "v"(ga)
                 : "memory");
    // W tile: 128 outs x 32 f32 -> f16 (conversion needs VALU anyway)
    const float* wg = W + (size_t)(n0 + lrow) * 512 + k0 + lseg;
#pragma unroll
    for (int e = 0; e < 16; ++e) Wsh[lrow * 32 + lseg + e] = (_Float16)wg[e];
    asm volatile("s_wait_asynccnt 0" ::: "memory");
    __syncthreads();

#pragma unroll
    for (int mi = 0; mi < 2; ++mi) {
      v16h a;
      const _Float16* ap = &Xs[(wm * 32 + mi * 16 + lanelo) * 32 + 16 * lane16];
#pragma unroll
      for (int e = 0; e < 16; ++e) a[e] = ap[e];
#pragma unroll
      for (int nj = 0; nj < 4; ++nj) {
        v16h b;
        const _Float16* bp =
            &Wsh[(wn * 64 + nj * 16 + lanelo) * 32 + 16 * lane16];
#pragma unroll
        for (int e = 0; e < 16; ++e) b[e] = bp[e];
        acc[mi][nj] = wmma_f16(a, b, acc[mi][nj]);
      }
    }
    __syncthreads();
  }

#pragma unroll
  for (int mi = 0; mi < 2; ++mi)
#pragma unroll
    for (int nj = 0; nj < 4; ++nj)
#pragma unroll
      for (int g = 0; g < 8; ++g) {
        const size_t m = m0 + wm * 32 + mi * 16 + g + 8 * lane16;
        const int    o = n0 + wn * 64 + nj * 16 + lanelo;
        Y[m * 512 + o] = acc[mi][nj][g];
      }
}

extern "C" void kernel_launch(void* const* d_in, const int* in_sizes, int n_in,
                              void* d_out, int out_size, void* d_ws,
                              size_t ws_size, hipStream_t stream) {
  (void)in_sizes; (void)n_in; (void)out_size; (void)ws_size;
  const float* x     = (const float*)d_in[0];
  const float* qw    = (const float*)d_in[1];
  const float* qb    = (const float*)d_in[2];
  const float* qg    = (const float*)d_in[3];
  const float* qbeta = (const float*)d_in[4];
  const float* kw    = (const float*)d_in[5];
  const float* kb    = (const float*)d_in[6];
  const float* kg    = (const float*)d_in[7];
  const float* kbeta = (const float*)d_in[8];
  const float* vw    = (const float*)d_in[9];
  const float* vb    = (const float*)d_in[10];
  const float* vg    = (const float*)d_in[11];
  const float* vbeta = (const float*)d_in[12];
  const float* relb  = (const float*)d_in[13];
  const float* outw  = (const float*)d_in[14];
  _Float16* attnX = (_Float16*)d_ws;  // (2048*49, 512) f16, ~103 MB (fits L2)
  float*    out   = (float*)d_out;

  const size_t lds1 = 100352 + 3 * 65536;  // 296,960 B <= 320 KB WGP LDS
  (void)hipFuncSetAttribute((const void*)attn_win_kernel,
                            hipFuncAttributeMaxDynamicSharedMemorySize,
                            (int)lds1);
  attn_win_kernel<<<2048, 256, lds1, stream>>>(x, qw, qb, qg, qbeta, kw, kb, kg,
                                               kbeta, vw, vb, vg, vbeta, relb,
                                               attnX);
  proj_kernel<<<dim3(784, 4), 256, 0, stream>>>(attnX, outw, out);
}
